// TransformerEncoderDecoder_73117523247345
// MI455X (gfx1250) — matrix operand order repaired and checked by reading, not compiled, not run on hardware
//
#include <hip/hip_runtime.h>
#include <hip/hip_bf16.h>

// ---------------------------------------------------------------------------
// CDNA5 (gfx1250) linear-attention transformer encoder-decoder forward.
// GEMMs + attention matrix einsums on v_wmma_f32_16x16x32_bf16 (wave32).
// ---------------------------------------------------------------------------

typedef __attribute__((ext_vector_type(16))) __bf16 v16bf;
typedef __attribute__((ext_vector_type(8)))  float  v8f;
typedef __attribute__((ext_vector_type(4)))  float  v4f;

#define DM     512
#define HNUM   8
#define DH     64
#define TSEQ   1024
#define BSZ    4
#define NTOK   4096      // B * T
#define FFDIM  2048
#define NCHUNK 8
#define CHUNKC 128

__device__ __forceinline__ float gelu_exact(float v) {
  return 0.5f * v * (1.f + erff(v * 0.70710678118654752f));
}

__device__ __forceinline__ v8f wmma_bf16(v16bf a, v16bf b, v8f c) {
  return __builtin_amdgcn_wmma_f32_16x16x32_bf16(false, a, false, b, (short)0, c,
                                                 false, false);
}

// A-fragment (16x32 bf16) from f32 row: p = row_base + k0 + half*16.
// elems 0..15 = K[k0+16*half .. +16).
__device__ __forceinline__ v16bf load_a_bf16(const float* p) {
  v4f x0 = *(const v4f*)(p);
  v4f x1 = *(const v4f*)(p + 4);
  v4f x2 = *(const v4f*)(p + 8);
  v4f x3 = *(const v4f*)(p + 12);
  v16bf r;
#pragma unroll
  for (int i = 0; i < 4; ++i) {
    r[i]      = (__bf16)x0[i];
    r[4 + i]  = (__bf16)x1[i];
    r[8 + i]  = (__bf16)x2[i];
    r[12 + i] = (__bf16)x3[i];
  }
  return r;
}

// B-fragment (32x16 bf16) from 16 contiguous f32: elems = p[0..15].
__device__ __forceinline__ v16bf load_b16_f32(const float* p) {
  v4f x0 = *(const v4f*)(p);
  v4f x1 = *(const v4f*)(p + 4);
  v4f x2 = *(const v4f*)(p + 8);
  v4f x3 = *(const v4f*)(p + 12);
  v16bf r;
#pragma unroll
  for (int i = 0; i < 4; ++i) {
    r[i]      = (__bf16)x0[i];
    r[4 + i]  = (__bf16)x1[i];
    r[8 + i]  = (__bf16)x2[i];
    r[12 + i] = (__bf16)x3[i];
  }
  return r;
}

// B-fragment from 16 contiguous bf16 (pre-transposed weights).
__device__ __forceinline__ v16bf load_b16_bf16(const __bf16* p) {
  return *(const v16bf*)p;
}

// ---------------------------------------------------------------------------
// WMMA GEMM: C[M,N] = epi(A[M,K] @ W[K,N] + bias), W pre-transposed bf16
// WT[N][K]. Wave computes WM x WN; block = 256 threads = 8 waves.
// EPI: 0 = identity, 1 = GELU, 2 = ReLU.
// ---------------------------------------------------------------------------
template<int BM, int BN, int WM, int WN, int EPI>
__global__ __launch_bounds__(256) void gemm_bf16_kernel(
    const float* __restrict__ A, const __bf16* __restrict__ WT,
    const float* __restrict__ bias, float* __restrict__ C,
    int M, int N, int K)
{
  (void)M;
  constexpr int WAVES_N = BN / WN;
  constexpr int MT = WM / 16, NT = WN / 16;
  const int wave = threadIdx.x >> 5;
  const int lane = threadIdx.x & 31;
  const int half = lane >> 4;
  const int l16  = lane & 15;
  const int m0 = blockIdx.y * BM + (wave / WAVES_N) * WM;
  const int n0 = blockIdx.x * BN + (wave % WAVES_N) * WN;

  v8f acc[MT][NT] = {};

  for (int k0 = 0; k0 < K; k0 += 32) {
    v16bf af[MT];
    v16bf bf[NT];
#pragma unroll
    for (int mi = 0; mi < MT; ++mi)
      af[mi] = load_a_bf16(A + (size_t)(m0 + mi * 16 + l16) * K + k0 + half * 16);
#pragma unroll
    for (int ni = 0; ni < NT; ++ni)
      bf[ni] = load_b16_bf16(WT + (size_t)(n0 + ni * 16 + l16) * K + k0 + half * 16);
#pragma unroll
    for (int mi = 0; mi < MT; ++mi)
#pragma unroll
      for (int ni = 0; ni < NT; ++ni)
        acc[mi][ni] = wmma_bf16(af[mi], bf[ni], acc[mi][ni]);
    if (k0 + 32 < K)  // prefetch next weight tile (global_prefetch_b8)
      __builtin_prefetch(WT + (size_t)(n0 + l16) * K + k0 + 32 + half * 16, 0, 1);
  }

#pragma unroll
  for (int mi = 0; mi < MT; ++mi)
#pragma unroll
    for (int ni = 0; ni < NT; ++ni) {
      const int col = n0 + ni * 16 + l16;
      const float bv = bias[col];
#pragma unroll
      for (int r = 0; r < 8; ++r) {
        const int row = m0 + mi * 16 + half * 8 + r;
        float v = acc[mi][ni][r] + bv;
        if (EPI == 1) v = gelu_exact(v);
        if (EPI == 2) v = fmaxf(v, 0.f);
        C[(size_t)row * N + col] = v;
      }
    }
}

// Convert W[K][N] f32 -> WT[N][K] bf16
__global__ void wconv_kernel(const float* __restrict__ W, __bf16* __restrict__ WT,
                             int K, int N) {
  int idx = blockIdx.x * 256 + threadIdx.x;
  if (idx < K * N) {
    int k = idx / N, n = idx - k * N;
    WT[(size_t)n * K + k] = (__bf16)W[idx];
  }
}

// phi(x) = elu(x) + 1 = x>0 ? x+1 : exp(x), in place
__global__ void phi_kernel(float* __restrict__ x, int n) {
  int i = blockIdx.x * 256 + threadIdx.x;
  if (i < n) {
    float v = x[i];
    x[i] = v > 0.f ? v + 1.f : __expf(v);
  }
}

// LayerNorm over last dim (512); optional residual r. One block per row.
__global__ __launch_bounds__(256) void ln_kernel(
    const float* __restrict__ x, const float* __restrict__ r,
    const float* __restrict__ g, const float* __restrict__ b,
    float* __restrict__ out)
{
  const int row = blockIdx.x, t = threadIdx.x;
  __shared__ float red[256];
  const size_t base = (size_t)row * DM;
  float v0 = x[base + t], v1 = x[base + t + 256];
  if (r) { v0 += r[base + t]; v1 += r[base + t + 256]; }
  red[t] = v0 + v1; __syncthreads();
  for (int off = 128; off; off >>= 1) { if (t < off) red[t] += red[t + off]; __syncthreads(); }
  const float mu = red[0] * (1.f / DM); __syncthreads();
  const float d0 = v0 - mu, d1 = v1 - mu;
  red[t] = d0 * d0 + d1 * d1; __syncthreads();
  for (int off = 128; off; off >>= 1) { if (t < off) red[t] += red[t + off]; __syncthreads(); }
  const float rstd = rsqrtf(red[0] * (1.f / DM) + 1e-5f);
  out[base + t]       = d0 * rstd * g[t]       + b[t];
  out[base + t + 256] = d1 * rstd * g[t + 256] + b[t + 256];
}

// Non-causal: KVT[b,h][m][d] = sum_s phiK[b,s,h,d] * V[b,s,h,m]; Ksum[b,h,d].
// Transposed output so downstream WMMA B-fragments are contiguous rows.
__global__ __launch_bounds__(256) void kv_kernel(
    const float* __restrict__ Kf, const float* __restrict__ Vf,
    float* __restrict__ KVT, float* __restrict__ Ksum)
{
  const int bh = blockIdx.x, b = bh >> 3, h = bh & 7;
  const int t = threadIdx.x;
  __shared__ float Ks[32][64];
  __shared__ float Vs[32][64];
  const int d = t >> 2, mb = (t & 3) << 4;
  float acc[16] = {};
  float ks = 0.f;
  const size_t base = ((size_t)b * TSEQ) * DM + h * DH;
  for (int s0 = 0; s0 < TSEQ; s0 += 32) {
#pragma unroll
    for (int i = 0; i < 8; ++i) {
      int e = t + i * 256, ss = e >> 6, dd = e & 63;
      Ks[ss][dd] = Kf[base + (size_t)(s0 + ss) * DM + dd];
      Vs[ss][dd] = Vf[base + (size_t)(s0 + ss) * DM + dd];
    }
    __syncthreads();
#pragma unroll 8
    for (int s = 0; s < 32; ++s) {
      float kv = Ks[s][d];
      if ((t & 3) == 0) ks += kv;
#pragma unroll
      for (int i = 0; i < 16; ++i) acc[i] += kv * Vs[s][mb + i];
    }
    __syncthreads();
  }
#pragma unroll
  for (int i = 0; i < 16; ++i)
    KVT[(size_t)bh * 4096 + (size_t)(mb + i) * DH + d] = acc[i];
  if ((t & 3) == 0) Ksum[bh * DH + d] = ks;
}

// WMMA non-causal output: out[tok, h*64+m] = (phiQ . KVT[m,:]) * rden(tok,h).
// Block per (b,h, 128-row slab); 8 waves x 16 rows.
__global__ __launch_bounds__(256) void linout_wmma_kernel(
    const float* __restrict__ Qf, const float* __restrict__ KVT,
    const float* __restrict__ Ksum, float* __restrict__ O)
{
  __shared__ float rden[128];
  const int blk = blockIdx.x;          // bh*8 + rb
  const int rb = blk & 7, bh = blk >> 3, b = bh >> 3, h = bh & 7;
  const int t = threadIdx.x;
  const int wave = t >> 5, lane = t & 31, half = lane >> 4, l16 = lane & 15;
  const size_t rowbase = ((size_t)b * TSEQ + rb * 128) * DM + h * DH;

  if (t < 128) {
    const float* qi = Qf + rowbase + (size_t)t * DM;
    const float* kp = Ksum + bh * DH;
    float s = 0.f;
#pragma unroll
    for (int d = 0; d < DH; ++d) s += qi[d] * kp[d];
    rden[t] = 1.f / (s + 1e-6f);
  }
  __syncthreads();

  const int i0 = wave * 16;
  v8f acc[4] = {};
  const float* kvt = KVT + (size_t)bh * 4096;
  for (int k0 = 0; k0 < DH; k0 += 32) {
    v16bf a = load_a_bf16(Qf + rowbase + (size_t)(i0 + l16) * DM + k0 + half * 16);
    v16bf bfr[4];
#pragma unroll
    for (int nt = 0; nt < 4; ++nt)
      bfr[nt] = load_b16_f32(kvt + (size_t)(nt * 16 + l16) * DH + k0 + half * 16);
#pragma unroll
    for (int nt = 0; nt < 4; ++nt) acc[nt] = wmma_bf16(a, bfr[nt], acc[nt]);
  }
#pragma unroll
  for (int nt = 0; nt < 4; ++nt) {
    const int m = nt * 16 + l16;
#pragma unroll
    for (int r = 0; r < 8; ++r) {
      const int i = i0 + half * 8 + r;
      O[rowbase + (size_t)i * DM + m] = acc[nt][r] * rden[i];
    }
  }
}

// Causal chunked: per-chunk KV ([d][m]) and Ksum. Block per (b,h,n).
__global__ __launch_bounds__(256) void chunk_kv_kernel(
    const float* __restrict__ Kf, const float* __restrict__ Vf,
    float* __restrict__ CKV, float* __restrict__ CKS)
{
  const int blk = blockIdx.x;
  const int n = blk & 7, bh = blk >> 3, b = bh >> 3, h = bh & 7;
  const int t = threadIdx.x;
  __shared__ float Ks[32][64];
  __shared__ float Vs[32][64];
  const int d = t >> 2, mb = (t & 3) << 4;
  float acc[16] = {};
  float ks = 0.f;
  const size_t base = ((size_t)b * TSEQ + n * CHUNKC) * DM + h * DH;
  for (int s0 = 0; s0 < CHUNKC; s0 += 32) {
#pragma unroll
    for (int i = 0; i < 8; ++i) {
      int e = t + i * 256, ss = e >> 6, dd = e & 63;
      Ks[ss][dd] = Kf[base + (size_t)(s0 + ss) * DM + dd];
      Vs[ss][dd] = Vf[base + (size_t)(s0 + ss) * DM + dd];
    }
    __syncthreads();
#pragma unroll 8
    for (int s = 0; s < 32; ++s) {
      float kv = Ks[s][d];
      if ((t & 3) == 0) ks += kv;
#pragma unroll
      for (int i = 0; i < 16; ++i) acc[i] += kv * Vs[s][mb + i];
    }
    __syncthreads();
  }
  float* outp = CKV + (size_t)blk * 4096 + d * 64 + mb;
#pragma unroll
  for (int i = 0; i < 16; ++i) outp[i] = acc[i];
  if ((t & 3) == 0) CKS[blk * DH + d] = ks;
}

// Exclusive prefix over the 8 chunks; writes the KV prefix TRANSPOSED
// (CKVPT[...][m][d]) so causal WMMA B-fragments are contiguous rows.
__global__ __launch_bounds__(256) void prefix_kernel(
    const float* __restrict__ CKV, const float* __restrict__ CKS,
    float* __restrict__ CKVPT, float* __restrict__ CKSP)
{
  const int bh = blockIdx.x, t = threadIdx.x;
  for (int e = t; e < 4096; e += 256) {
    int d = e >> 6, m = e & 63;
    float run = 0.f;
#pragma unroll
    for (int n = 0; n < NCHUNK; ++n) {
      size_t srci = ((size_t)bh * NCHUNK + n) * 4096 + e;              // [d][m]
      size_t dsti = ((size_t)bh * NCHUNK + n) * 4096 + m * DH + d;     // [m][d]
      CKVPT[dsti] = run;
      run += CKV[srci];
    }
  }
  if (t < DH) {
    float run = 0.f;
#pragma unroll
    for (int n = 0; n < NCHUNK; ++n) {
      size_t idx = ((size_t)bh * NCHUNK + n) * DH + t;
      CKSP[idx] = run;
      run += CKS[idx];
    }
  }
}

// WMMA intra-chunk causal attention. Block per (b,h,n); 8 waves.
// LDS: sc[128][128] scores, vt[64][128] = Vc^T, rden[128].
__global__ __launch_bounds__(256) void causal_wmma_kernel(
    const float* __restrict__ Qf, const float* __restrict__ Kf,
    const float* __restrict__ Vf, const float* __restrict__ CKVPT,
    const float* __restrict__ CKSP, float* __restrict__ O)
{
  extern __shared__ float sm[];
  float* sc   = sm;                 // 128*128
  float* vt   = sm + 128 * 128;     // 64*128
  float* rden = vt + 64 * 128;      // 128
  const int blk = blockIdx.x;
  const int n = blk & 7, bh = blk >> 3, b = bh >> 3, h = bh & 7;
  const int t = threadIdx.x;
  const int wave = t >> 5, lane = t & 31, half = lane >> 4, l16 = lane & 15;
  const size_t rowbase = ((size_t)b * TSEQ + n * CHUNKC) * DM + h * DH;
  const int i0 = wave * 16;

  // stage Vc^T into LDS
  for (int e = t; e < 128 * 64; e += 256) {
    int j = e >> 6, m = e & 63;
    vt[m * 128 + j] = Vf[rowbase + (size_t)j * DM + m];
  }

  // phase 1: S = tril(Qc Kc^T); wave computes 16x128 strip
  {
    v8f acc[8] = {};
    for (int k0 = 0; k0 < DH; k0 += 32) {
      v16bf a = load_a_bf16(Qf + rowbase + (size_t)(i0 + l16) * DM + k0 + half * 16);
      v16bf bfr[8];
#pragma unroll
      for (int nt = 0; nt < 8; ++nt)
        bfr[nt] = load_b16_f32(Kf + rowbase + (size_t)(nt * 16 + l16) * DM + k0 + half * 16);
#pragma unroll
      for (int nt = 0; nt < 8; ++nt) acc[nt] = wmma_bf16(a, bfr[nt], acc[nt]);
    }
#pragma unroll
    for (int nt = 0; nt < 8; ++nt) {
      const int j = nt * 16 + l16;
#pragma unroll
      for (int r = 0; r < 8; ++r) {
        const int i = i0 + half * 8 + r;
        sc[i * 128 + j] = (j <= i) ? acc[nt][r] : 0.f;
      }
    }
  }
  __syncthreads();

  // denominators
  if (t < 128) {
    const float* qi = Qf + rowbase + (size_t)t * DM;
    const float* ksp = CKSP + ((size_t)bh * NCHUNK + n) * DH;
    float den = 0.f;
    for (int j = 0; j < 128; ++j) den += sc[t * 128 + j];
#pragma unroll
    for (int d = 0; d < DH; ++d) den += qi[d] * ksp[d];
    rden[t] = 1.f / (den + 1e-6f);
  }
  __syncthreads();

  // phase 2: num = S @ Vc + Qc @ kv_prefix^T ; out = num * rden
  {
    v8f acc[4] = {};
    for (int k0 = 0; k0 < 128; k0 += 32) {           // S @ Vc (LDS operands)
      v16bf a = load_a_bf16(sc + (size_t)(i0 + l16) * 128 + k0 + half * 16);
      v16bf bfr[4];
#pragma unroll
      for (int nt = 0; nt < 4; ++nt)
        bfr[nt] = load_b16_f32(vt + (size_t)(nt * 16 + l16) * 128 + k0 + half * 16);
#pragma unroll
      for (int nt = 0; nt < 4; ++nt) acc[nt] = wmma_bf16(a, bfr[nt], acc[nt]);
    }
    const float* kvpT = CKVPT + ((size_t)bh * NCHUNK + n) * 4096;
    for (int k0 = 0; k0 < DH; k0 += 32) {            // Qc @ kvp^T
      v16bf a = load_a_bf16(Qf + rowbase + (size_t)(i0 + l16) * DM + k0 + half * 16);
      v16bf bfr[4];
#pragma unroll
      for (int nt = 0; nt < 4; ++nt)
        bfr[nt] = load_b16_f32(kvpT + (size_t)(nt * 16 + l16) * DH + k0 + half * 16);
#pragma unroll
      for (int nt = 0; nt < 4; ++nt) acc[nt] = wmma_bf16(a, bfr[nt], acc[nt]);
    }
#pragma unroll
    for (int nt = 0; nt < 4; ++nt) {
      const int m = nt * 16 + l16;
#pragma unroll
      for (int r = 0; r < 8; ++r) {
        const int i = i0 + half * 8 + r;
        O[rowbase + (size_t)i * DM + m] = acc[nt][r] * rden[i];
      }
    }
  }
}

// ---------------------------------------------------------------------------
// Host orchestration
// ---------------------------------------------------------------------------
extern "C" void kernel_launch(void* const* d_in, const int* in_sizes, int n_in,
                              void* d_out, int out_size, void* d_ws, size_t ws_size,
                              hipStream_t stream)
{
  (void)in_sizes; (void)n_in; (void)out_size; (void)ws_size;
  // d_in[0]=src, d_in[1]=tgt, d_in[2]=tgt_mask (unused), then params leaves
  // flattened with sorted dict keys (JAX tree order).
  const int P = 3;
  auto in = [&](int i) { return (const float*)d_in[i]; };
  const float* src = in(0);
  const float* tgt = in(1);

  float* f = (float*)d_ws;
  size_t fo = 0;
  auto AF = [&](size_t nelem) { float* p = f + fo; fo += (nelem + 7) & ~(size_t)7; return p; };
  float* xbuf   = AF((size_t)NTOK * DM);
  float* ybuf   = AF((size_t)NTOK * DM);
  float* membuf = AF((size_t)NTOK * DM);
  float* tbuf   = AF((size_t)NTOK * DM);
  float* t2buf  = AF((size_t)NTOK * DM);
  float* qbuf   = AF((size_t)NTOK * DM);
  float* kbuf   = AF((size_t)NTOK * DM);
  float* vbuf   = AF((size_t)NTOK * DM);
  float* hidden = AF((size_t)NTOK * FFDIM);
  float* kvtbuf = AF((size_t)BSZ * HNUM * DH * DH);
  float* ksum   = AF((size_t)BSZ * HNUM * DH);
  float* ckv    = AF((size_t)BSZ * HNUM * NCHUNK * DH * DH);
  float* ckvpt  = AF((size_t)BSZ * HNUM * NCHUNK * DH * DH);
  float* cks    = AF((size_t)BSZ * HNUM * NCHUNK * DH);
  float* cksp   = AF((size_t)BSZ * HNUM * NCHUNK * DH);

  __bf16* wt = (__bf16*)(f + fo);
  size_t wo = 0;
  auto conv = [&](int idx, int K, int N) -> __bf16* {
    __bf16* dst = wt + wo;
    wo += (size_t)K * N;
    int nelem = K * N;
    wconv_kernel<<<dim3((nelem + 255) / 256), dim3(256), 0, stream>>>(in(idx), dst, K, N);
    return dst;
  };

  __bf16* srcWT = conv(P + 182, 64, DM);
  __bf16* tgtWT = conv(P + 184, 64, DM);
  __bf16* finWT = conv(P + 180, DM, 64);
  __bf16 *eWq[4], *eWk[4], *eWv[4], *eWo[4], *eW1[4], *eW2[4];
  for (int l = 0; l < 4; ++l) {
    int EL = P + 106 + 18 * l;
    eWk[l] = conv(EL + 0, DM, DM);
    eWo[l] = conv(EL + 1, DM, DM);
    eWq[l] = conv(EL + 2, DM, DM);
    eWv[l] = conv(EL + 3, DM, DM);
    eW1[l] = conv(EL + 8, DM, FFDIM);
    eW2[l] = conv(EL + 9, FFDIM, DM);
  }
  __bf16 *dsWq[4], *dsWk[4], *dsWv[4], *dsWo[4];
  __bf16 *dcWq[4], *dcWk[4], *dcWv[4], *dcWo[4], *dW1[4], *dW2[4];
  for (int l = 0; l < 4; ++l) {
    int DL = P + 26 * l;
    dcWk[l] = conv(DL + 0, DM, DM);
    dcWo[l] = conv(DL + 1, DM, DM);
    dcWq[l] = conv(DL + 2, DM, DM);
    dcWv[l] = conv(DL + 3, DM, DM);
    dW1[l]  = conv(DL + 8, DM, FFDIM);
    dW2[l]  = conv(DL + 9, FFDIM, DM);
    dsWk[l] = conv(DL + 18, DM, DM);
    dsWo[l] = conv(DL + 19, DM, DM);
    dsWq[l] = conv(DL + 20, DM, DM);
    dsWv[l] = conv(DL + 21, DM, DM);
  }

  auto gemm = [&](const float* A, const __bf16* W, const float* bias, float* C,
                  int N, int K, int epi) {
    dim3 g(N / 128, NTOK / 256), blk(256);
    if (epi == 0)
      gemm_bf16_kernel<256, 128, 64, 64, 0><<<g, blk, 0, stream>>>(A, W, bias, C, NTOK, N, K);
    else if (epi == 1)
      gemm_bf16_kernel<256, 128, 64, 64, 1><<<g, blk, 0, stream>>>(A, W, bias, C, NTOK, N, K);
    else
      gemm_bf16_kernel<256, 128, 64, 64, 2><<<g, blk, 0, stream>>>(A, W, bias, C, NTOK, N, K);
  };
  auto phi = [&](float* p) {
    phi_kernel<<<dim3((NTOK * DM) / 256), dim3(256), 0, stream>>>(p, NTOK * DM);
  };
  auto ln = [&](const float* x, const float* r, int gi, int bi, float* outp) {
    ln_kernel<<<dim3(NTOK), dim3(256), 0, stream>>>(x, r, in(gi), in(bi), outp);
  };
  const size_t causal_lds = (128 * 128 + 64 * 128 + 128) * sizeof(float);

  // ---------------- encoder ----------------
  gemm(src, srcWT, in(P + 183), xbuf, DM, 64, 1);
  for (int l = 0; l < 4; ++l) {
    int EL = P + 106 + 18 * l;
    gemm(xbuf, eWq[l], in(EL + 6), qbuf, DM, DM, 0);
    gemm(xbuf, eWk[l], in(EL + 4), kbuf, DM, DM, 0);
    gemm(xbuf, eWv[l], in(EL + 7), vbuf, DM, DM, 0);
    phi(qbuf); phi(kbuf);
    kv_kernel<<<dim3(BSZ * HNUM), dim3(256), 0, stream>>>(kbuf, vbuf, kvtbuf, ksum);
    linout_wmma_kernel<<<dim3(BSZ * HNUM * 8), dim3(256), 0, stream>>>(qbuf, kvtbuf, ksum, tbuf);
    gemm(tbuf, eWo[l], in(EL + 5), t2buf, DM, DM, 0);
    ln(xbuf, t2buf, EL + 13, EL + 12, xbuf);
    gemm(xbuf, eW1[l], in(EL + 10), hidden, FFDIM, DM, 2);
    gemm(hidden, eW2[l], in(EL + 11), t2buf, DM, FFDIM, 0);
    ln(xbuf, t2buf, EL + 15, EL + 14, xbuf);
  }
  ln(xbuf, nullptr, P + 179, P + 178, membuf);

  // ---------------- decoder ----------------
  gemm(tgt, tgtWT, in(P + 185), ybuf, DM, 64, 1);
  for (int l = 0; l < 4; ++l) {
    int DL = P + 26 * l;
    // causal self-attention
    gemm(ybuf, dsWq[l], in(DL + 24), qbuf, DM, DM, 0);
    gemm(ybuf, dsWk[l], in(DL + 22), kbuf, DM, DM, 0);
    gemm(ybuf, dsWv[l], in(DL + 25), vbuf, DM, DM, 0);
    phi(qbuf); phi(kbuf);
    chunk_kv_kernel<<<dim3(BSZ * HNUM * NCHUNK), dim3(256), 0, stream>>>(kbuf, vbuf, ckv, cks);
    prefix_kernel<<<dim3(BSZ * HNUM), dim3(256), 0, stream>>>(ckv, cks, ckvpt, cksp);
    causal_wmma_kernel<<<dim3(BSZ * HNUM * NCHUNK), dim3(256), causal_lds, stream>>>(
        qbuf, kbuf, vbuf, ckvpt, cksp, tbuf);
    gemm(tbuf, dsWo[l], in(DL + 23), t2buf, DM, DM, 0);
    ln(ybuf, t2buf, DL + 13, DL + 12, ybuf);
    // cross-attention
    gemm(ybuf, dcWq[l], in(DL + 6), qbuf, DM, DM, 0);
    gemm(membuf, dcWk[l], in(DL + 4), kbuf, DM, DM, 0);
    gemm(membuf, dcWv[l], in(DL + 7), vbuf, DM, DM, 0);
    phi(qbuf); phi(kbuf);
    kv_kernel<<<dim3(BSZ * HNUM), dim3(256), 0, stream>>>(kbuf, vbuf, kvtbuf, ksum);
    linout_wmma_kernel<<<dim3(BSZ * HNUM * 8), dim3(256), 0, stream>>>(qbuf, kvtbuf, ksum, tbuf);
    gemm(tbuf, dcWo[l], in(DL + 5), t2buf, DM, DM, 0);
    ln(ybuf, t2buf, DL + 15, DL + 14, ybuf);
    // FFN
    gemm(ybuf, dW1[l], in(DL + 10), hidden, FFDIM, DM, 2);
    gemm(hidden, dW2[l], in(DL + 11), t2buf, DM, FFDIM, 0);
    ln(ybuf, t2buf, DL + 17, DL + 16, ybuf);
  }
  ln(ybuf, nullptr, P + 105, P + 104, tbuf);

  // final projection: gelu(y @ fin_W + fin_b) -> d_out, N = 64
  {
    dim3 g(1, NTOK / 128), blk(256);
    gemm_bf16_kernel<128, 64, 32, 32, 1><<<g, blk, 0, stream>>>(
        tbuf, finWT, in(P + 181), (float*)d_out, NTOK, 64, DM);
  }
}
